// VRNN_ss_8847632630167
// MI455X (gfx1250) — compile-verified, matrix operand order checked by reading, not run on hardware
//
#include <hip/hip_runtime.h>
#include <hip/hip_bf16.h>

// ---------------------------------------------------------------------------
// VRNN forward on MI455X (gfx1250), wave32 + v_wmma_f32_16x16x32_bf16.
//  Phase A: transpose/convert all weights fp32 -> bf16 [N][K] layout.
//  Phase B: fx = dense2(x) for all T*B rows (fully parallel WMMA GEMM).
//  Phase C: persistent recurrence kernel: 16 WGs x 16 batch rows, t-loop
//           inside, all GEMMs as 16x16 WMMA tiles over bf16 LDS activations.
//  This round: native gfx1250 transcendentals (v_tanh_f32 / v_exp_f32 /
//  v_rcp_f32) replace the branchy ocml tanh/exp expansions so activations
//  co-execute with the WMMA (XDL) stream instead of serializing it.
// ---------------------------------------------------------------------------

typedef __attribute__((ext_vector_type(16))) __bf16 v16bf;
typedef __attribute__((ext_vector_type(8)))  __bf16 v8bf;
typedef __attribute__((ext_vector_type(8)))  float  v8f;

#define TT 256
#define BB 256
#define LDA 520   // 512 + 8 halves padding (bank spread), 16B-aligned rows
#define LDZ 72    // 64 + 8 halves padding

// ---- workspace layout (bf16 elements) -------------------------------------
constexpr size_t O_FX1   = 0;
constexpr size_t O_FX2   = O_FX1   + (size_t)512 * 64;
constexpr size_t O_FZ1   = O_FX2   + (size_t)512 * 512;
constexpr size_t O_FZ2   = O_FZ1   + (size_t)512 * 64;
constexpr size_t O_INF1  = O_FZ2   + (size_t)512 * 512;
constexpr size_t O_INF2  = O_INF1  + (size_t)512 * 1024;
constexpr size_t O_INFMU = O_INF2  + (size_t)512 * 512;
constexpr size_t O_INFLV = O_INFMU + (size_t)64 * 512;
constexpr size_t O_PRI1  = O_INFLV + (size_t)64 * 512;
constexpr size_t O_PRI2  = O_PRI1  + (size_t)512 * 512;
constexpr size_t O_PRIMU = O_PRI2  + (size_t)512 * 512;
constexpr size_t O_PRILV = O_PRIMU + (size_t)64 * 512;
constexpr size_t O_GEN1  = O_PRILV + (size_t)64 * 512;
constexpr size_t O_GEN2  = O_GEN1  + (size_t)512 * 1024;
constexpr size_t O_GENMU = O_GEN2  + (size_t)512 * 512;
constexpr size_t O_LSTM  = O_GENMU + (size_t)64 * 512;        // [2048][1536]
constexpr size_t O_FXG   = O_LSTM  + (size_t)2048 * 1536;     // [T*B][512]
constexpr size_t N_BF16  = O_FXG   + (size_t)TT * BB * 512;
constexpr size_t CWS_BYTE = ((N_BF16 * 2 + 255) / 256) * 256; // f32 c-state

// ---- fast native transcendentals (gfx1250) ---------------------------------
__device__ inline float fast_tanh(float x) {
#if __has_builtin(__builtin_amdgcn_tanhf)
  return __builtin_amdgcn_tanhf(x);
#else
  return tanhf(x);
#endif
}
__device__ inline float fast_exp2(float x) {
#if __has_builtin(__builtin_amdgcn_exp2f)
  return __builtin_amdgcn_exp2f(x);
#else
  return exp2f(x);
#endif
}
__device__ inline float fast_rcp(float x) {
#if __has_builtin(__builtin_amdgcn_rcpf)
  return __builtin_amdgcn_rcpf(x);
#else
  return 1.f / x;
#endif
}
#define LOG2E 1.44269504088896340736f
__device__ inline float fast_exp(float x) { return fast_exp2(x * LOG2E); }
__device__ inline float sigf(float x) { return fast_rcp(1.f + fast_exp2(-LOG2E * x)); }

// ---- helpers ---------------------------------------------------------------
__device__ inline v16bf ldsA(const __bf16* p) {
  union { v16bf v; v8bf h[2]; } u;
  u.h[0] = *(const v8bf*)(p);
  u.h[1] = *(const v8bf*)(p + 8);
  return u.v;
}

// Accumulate one K-segment into a 16x16 f32 tile.
// A: LDS activations, row stride lda, this wave reads rows 0..15.
// wrow: per-lane weight row pointer = Wt + (n0+idx)*Ktot + kofs.
__device__ inline v8f acc_seg(v8f acc, const __bf16* A, int lda, int K,
                              const __bf16* wrow) {
  const int lane = threadIdx.x & 31;
  const int half = lane >> 4;   // 0: K 0..15,  1: K 16..31 of each step
  const int idx  = lane & 15;   // A: row m     B: column n
  const __bf16* ap = A + idx * lda + half * 16;
  const __bf16* bp = wrow + half * 16;
#pragma unroll 4
  for (int k = 0; k < K; k += 32) {
    v16bf a = ldsA(ap + k);
    v16bf b = *(const v16bf*)(bp + k);
    acc = __builtin_amdgcn_wmma_f32_16x16x32_bf16(false, a, false, b,
                                                  (short)0, acc, false, false);
  }
  return acc;
}

__device__ inline void st_bias_tanh(v8f a, const float* bias, int n0,
                                    __bf16* Out, int lda) {
  const int lane = threadIdx.x & 31;
  const int half = lane >> 4, idx = lane & 15;
  float bv = bias[n0 + idx];
#pragma unroll
  for (int r = 0; r < 8; ++r)
    Out[(r + half * 8) * lda + n0 + idx] = (__bf16)fast_tanh(a[r] + bv);
}

__device__ inline void st_head(v8f a, const float* bias, int n0,
                               float* Out, size_t rowbase, float* ldsbuf) {
  const int lane = threadIdx.x & 31;
  const int half = lane >> 4, idx = lane & 15;
  float bv = bias[n0 + idx];
#pragma unroll
  for (int r = 0; r < 8; ++r) {
    int m = r + half * 8;
    float v = a[r] + bv;
    Out[(rowbase + m) * 64 + n0 + idx] = v;
    if (ldsbuf) ldsbuf[m * LDZ + n0 + idx] = v;
  }
}

// ---- Phase A: fp32 [K][N] -> bf16 [N][ldo] transpose ----------------------
__global__ void kconvT(const float* __restrict__ in, __bf16* __restrict__ out,
                       int K, int N, int ldo, int kofs) {
  int i = blockIdx.x * blockDim.x + threadIdx.x;
  if (i >= K * N) return;
  int k = i / N, n = i % N;
  out[(size_t)n * ldo + kofs + k] = (__bf16)in[i];
}

// ---- Phase B: fx = tanh(tanh(x@W1+b1)@W2+b2) for all T*B rows -------------
__global__ __launch_bounds__(1024, 1)
void fx_kernel(const float* __restrict__ x, const float* __restrict__ b1,
               const float* __restrict__ b2, const __bf16* __restrict__ wt,
               __bf16* __restrict__ fxg) {
  __shared__ __bf16 sX[16 * LDZ];
  __shared__ __bf16 sT[16 * LDA];
  const int tid = threadIdx.x;
  const int wave = tid >> 5, lane = tid & 31;
  const int half = lane >> 4, idx = lane & 15;
  const size_t r0 = (size_t)blockIdx.x * 16;

  if (tid < 16 * 64) {
    int m = tid >> 6, n = tid & 63;
    sX[m * LDZ + n] = (__bf16)x[(r0 + m) * 64 + n];
  }
  __syncthreads();
  {
    int n0 = wave * 16;
    v8f a = {};
    a = acc_seg(a, sX, LDZ, 64, wt + O_FX1 + (size_t)(n0 + idx) * 64);
    st_bias_tanh(a, b1, n0, sT, LDA);
  }
  __syncthreads();
  {
    int n0 = wave * 16;
    v8f a = {};
    a = acc_seg(a, sT, LDA, 512, wt + O_FX2 + (size_t)(n0 + idx) * 512);
    float bv = b2[n0 + idx];
#pragma unroll
    for (int r = 0; r < 8; ++r)
      fxg[(r0 + r + half * 8) * 512 + n0 + idx] = (__bf16)fast_tanh(a[r] + bv);
  }
}

// ---- Phase C: persistent recurrence ---------------------------------------
__global__ __launch_bounds__(1024, 1)
void vrnn_step_kernel(const float* __restrict__ eps,
                      const float* b_inf1, const float* b_inf2,
                      const float* b_infmu, const float* b_inflv,
                      const float* b_fz1, const float* b_fz2,
                      const float* b_pri1, const float* b_pri2,
                      const float* b_primu, const float* b_prilv,
                      const float* b_gen1, const float* b_gen2,
                      const float* b_genmu,
                      const float* b_bih, const float* b_bhh,
                      const __bf16* __restrict__ wt,
                      float* __restrict__ cws,
                      float* __restrict__ out) {
  __shared__ __bf16 sFx[16 * LDA], sFz[16 * LDA], sH[16 * LDA];
  __shared__ __bf16 sTA[16 * LDA], sTB[16 * LDA];
  __shared__ float  sMu[16 * LDZ], sLv[16 * LDZ];
  __shared__ __bf16 sZ[16 * LDZ];

  const int tid = threadIdx.x;
  const int wave = tid >> 5, lane = tid & 31;
  const int half = lane >> 4, idx = lane & 15;
  const int bbase = blockIdx.x * 16;

  const size_t OS = (size_t)TT * BB * 64;
  float* oRec = out;            float* oZ   = out + OS;
  float* oMuQ = out + 2 * OS;   float* oLvQ = out + 3 * OS;
  float* oMuP = out + 4 * OS;   float* oLvP = out + 5 * OS;
  const __bf16* FXG = wt + O_FXG;
  const __bf16* WL  = wt + O_LSTM;

  // h0 = c0 = 0
  for (int i = tid; i < 16 * LDA; i += 1024) sH[i] = (__bf16)0.f;
  for (int i = tid; i < 16 * 512; i += 1024)
    cws[((size_t)bbase + (i >> 9)) * 512 + (i & 511)] = 0.f;
  __syncthreads();

  for (int t = 0; t < TT; ++t) {
    const size_t rowbase = (size_t)t * BB + bbase;

    // fetch precomputed fx tile (coalesced 16B per thread)
    {
      int m = tid >> 6, c0 = (tid & 63) * 8;
      *(uint4*)&sFx[m * LDA + c0] = *(const uint4*)(FXG + (rowbase + m) * 512 + c0);
      if (t + 1 < TT)
        __builtin_prefetch(FXG + (rowbase + BB + m) * 512 + c0, 0, 1);
    }
    __syncthreads();

    // inference l1: tanh([fx,h] @ inf_w1 + b1) -> sTA
    {
      int n0 = wave * 16;
      const __bf16* wr = wt + O_INF1 + (size_t)(n0 + idx) * 1024;
      v8f a = {};
      a = acc_seg(a, sFx, LDA, 512, wr);
      a = acc_seg(a, sH,  LDA, 512, wr + 512);
      st_bias_tanh(a, b_inf1, n0, sTA, LDA);
    }
    __syncthreads();

    // inference l2 -> hid_q in sTB
    {
      int n0 = wave * 16;
      v8f a = {};
      a = acc_seg(a, sTA, LDA, 512, wt + O_INF2 + (size_t)(n0 + idx) * 512);
      st_bias_tanh(a, b_inf2, n0, sTB, LDA);
    }
    __syncthreads();

    // heads mu_q / lv_q (to d_out + LDS staging)
    if (wave < 4) {
      int n0 = wave * 16;
      v8f a = {};
      a = acc_seg(a, sTB, LDA, 512, wt + O_INFMU + (size_t)(n0 + idx) * 512);
      st_head(a, b_infmu, n0, oMuQ, rowbase, sMu);
    } else if (wave < 8) {
      int n0 = (wave - 4) * 16;
      v8f a = {};
      a = acc_seg(a, sTB, LDA, 512, wt + O_INFLV + (size_t)(n0 + idx) * 512);
      st_head(a, b_inflv, n0, oLvQ, rowbase, sLv);
    }
    __syncthreads();

    // z = mu_q + eps * exp(0.5*lv_q)
    if (tid < 16 * 64) {
      int m = tid >> 6, n = tid & 63;
      float mu = sMu[m * LDZ + n], lv = sLv[m * LDZ + n];
      float z = mu + eps[(rowbase + m) * 64 + n] * fast_exp(0.5f * lv);
      oZ[(rowbase + m) * 64 + n] = z;
      sZ[m * LDZ + n] = (__bf16)z;
    }
    __syncthreads();

    // fz l1: tanh(z @ fz_w1 + b1) -> sTA
    {
      int n0 = wave * 16;
      v8f a = {};
      a = acc_seg(a, sZ, LDZ, 64, wt + O_FZ1 + (size_t)(n0 + idx) * 64);
      st_bias_tanh(a, b_fz1, n0, sTA, LDA);
    }
    __syncthreads();

    // fz l2 -> sFz
    {
      int n0 = wave * 16;
      v8f a = {};
      a = acc_seg(a, sTA, LDA, 512, wt + O_FZ2 + (size_t)(n0 + idx) * 512);
      st_bias_tanh(a, b_fz2, n0, sFz, LDA);
    }
    __syncthreads();

    // prior l1: tanh(h @ pri_w1 + b1) -> sTA
    {
      int n0 = wave * 16;
      v8f a = {};
      a = acc_seg(a, sH, LDA, 512, wt + O_PRI1 + (size_t)(n0 + idx) * 512);
      st_bias_tanh(a, b_pri1, n0, sTA, LDA);
    }
    __syncthreads();

    // prior l2 -> hid_p in sTB
    {
      int n0 = wave * 16;
      v8f a = {};
      a = acc_seg(a, sTA, LDA, 512, wt + O_PRI2 + (size_t)(n0 + idx) * 512);
      st_bias_tanh(a, b_pri2, n0, sTB, LDA);
    }
    __syncthreads();

    // heads mu_p / lv_p  (waves 0..7)  +  generator l1 (all waves)
    if (wave < 4) {
      int n0 = wave * 16;
      v8f a = {};
      a = acc_seg(a, sTB, LDA, 512, wt + O_PRIMU + (size_t)(n0 + idx) * 512);
      st_head(a, b_primu, n0, oMuP, rowbase, nullptr);
    } else if (wave < 8) {
      int n0 = (wave - 4) * 16;
      v8f a = {};
      a = acc_seg(a, sTB, LDA, 512, wt + O_PRILV + (size_t)(n0 + idx) * 512);
      st_head(a, b_prilv, n0, oLvP, rowbase, nullptr);
    }
    {
      int n0 = wave * 16;
      const __bf16* wr = wt + O_GEN1 + (size_t)(n0 + idx) * 1024;
      v8f a = {};
      a = acc_seg(a, sFz, LDA, 512, wr);
      a = acc_seg(a, sH,  LDA, 512, wr + 512);
      st_bias_tanh(a, b_gen1, n0, sTA, LDA);
    }
    __syncthreads();

    // generator l2 -> hid_g in sTB
    {
      int n0 = wave * 16;
      v8f a = {};
      a = acc_seg(a, sTA, LDA, 512, wt + O_GEN2 + (size_t)(n0 + idx) * 512);
      st_bias_tanh(a, b_gen2, n0, sTB, LDA);
    }
    __syncthreads();

    // recon head (waves 0..3) + LSTM gate GEMMs (all waves, regs)
    if (wave < 4) {
      int n0 = wave * 16;
      v8f a = {};
      a = acc_seg(a, sTB, LDA, 512, wt + O_GENMU + (size_t)(n0 + idx) * 512);
      st_head(a, b_genmu, n0, oRec, rowbase, nullptr);
    }
    v8f gi = {}, gf = {}, gg = {}, go = {};
    {
      int n0 = wave * 16;
#pragma unroll
      for (int g = 0; g < 4; ++g) {
        const __bf16* wr = WL + (size_t)(g * 512 + n0 + idx) * 1536;
        v8f a = {};
        a = acc_seg(a, sFx, LDA, 512, wr);
        a = acc_seg(a, sFz, LDA, 512, wr + 512);
        a = acc_seg(a, sH,  LDA, 512, wr + 1024);
        if (g == 0) gi = a; else if (g == 1) gf = a;
        else if (g == 2) gg = a; else go = a;
      }
    }
    __syncthreads();   // all reads of old h complete

    // LSTM cell update; write c (global f32) and h_new (LDS bf16)
    {
      int n0 = wave * 16;
      int n = n0 + idx;
      float bI = b_bih[n]        + b_bhh[n];
      float bF = b_bih[512 + n]  + b_bhh[512 + n];
      float bG = b_bih[1024 + n] + b_bhh[1024 + n];
      float bO = b_bih[1536 + n] + b_bhh[1536 + n];
#pragma unroll
      for (int r = 0; r < 8; ++r) {
        int m = r + half * 8;
        size_t ci = ((size_t)bbase + m) * 512 + n;
        float iv = sigf(gi[r] + bI);
        float fv = sigf(gf[r] + bF);
        float gv = fast_tanh(gg[r] + bG);
        float ov = sigf(go[r] + bO);
        float cn = fv * cws[ci] + iv * gv;
        cws[ci] = cn;
        sH[m * LDA + n] = (__bf16)(ov * fast_tanh(cn));
      }
    }
    __syncthreads();
  }
}

// ---------------------------------------------------------------------------
extern "C" void kernel_launch(void* const* d_in, const int* in_sizes, int n_in,
                              void* d_out, int out_size, void* d_ws, size_t ws_size,
                              hipStream_t stream) {
  (void)in_sizes; (void)n_in; (void)out_size; (void)ws_size;
  const float* x        = (const float*)d_in[0];
  const float* eps      = (const float*)d_in[1];
  const float* fx_w1    = (const float*)d_in[2];
  const float* fx_b1    = (const float*)d_in[3];
  const float* fx_w2    = (const float*)d_in[4];
  const float* fx_b2    = (const float*)d_in[5];
  const float* fz_w1    = (const float*)d_in[6];
  const float* fz_b1    = (const float*)d_in[7];
  const float* fz_w2    = (const float*)d_in[8];
  const float* fz_b2    = (const float*)d_in[9];
  const float* inf_w1   = (const float*)d_in[10];
  const float* inf_b1   = (const float*)d_in[11];
  const float* inf_w2   = (const float*)d_in[12];
  const float* inf_b2   = (const float*)d_in[13];
  const float* inf_mu_w = (const float*)d_in[14];
  const float* inf_mu_b = (const float*)d_in[15];
  const float* inf_lv_w = (const float*)d_in[16];
  const float* inf_lv_b = (const float*)d_in[17];
  const float* pri_w1   = (const float*)d_in[18];
  const float* pri_b1   = (const float*)d_in[19];
  const float* pri_w2   = (const float*)d_in[20];
  const float* pri_b2   = (const float*)d_in[21];
  const float* pri_mu_w = (const float*)d_in[22];
  const float* pri_mu_b = (const float*)d_in[23];
  const float* pri_lv_w = (const float*)d_in[24];
  const float* pri_lv_b = (const float*)d_in[25];
  const float* gen_w1   = (const float*)d_in[26];
  const float* gen_b1   = (const float*)d_in[27];
  const float* gen_w2   = (const float*)d_in[28];
  const float* gen_b2   = (const float*)d_in[29];
  const float* gen_mu_w = (const float*)d_in[30];
  const float* gen_mu_b = (const float*)d_in[31];
  const float* lstm_wih = (const float*)d_in[32];
  const float* lstm_bih = (const float*)d_in[33];
  const float* lstm_whh = (const float*)d_in[34];
  const float* lstm_bhh = (const float*)d_in[35];

  __bf16* wsb = (__bf16*)d_ws;
  float*  cws = (float*)((char*)d_ws + CWS_BYTE);

  auto TRS = [&](const float* in, size_t off, int K, int N, int ldo, int kofs) {
    int total = K * N;
    kconvT<<<(total + 255) / 256, 256, 0, stream>>>(in, wsb + off, K, N, ldo, kofs);
  };
  TRS(fx_w1,    O_FX1,   64,   512,  64,   0);
  TRS(fx_w2,    O_FX2,   512,  512,  512,  0);
  TRS(fz_w1,    O_FZ1,   64,   512,  64,   0);
  TRS(fz_w2,    O_FZ2,   512,  512,  512,  0);
  TRS(inf_w1,   O_INF1,  1024, 512,  1024, 0);
  TRS(inf_w2,   O_INF2,  512,  512,  512,  0);
  TRS(inf_mu_w, O_INFMU, 512,  64,   512,  0);
  TRS(inf_lv_w, O_INFLV, 512,  64,   512,  0);
  TRS(pri_w1,   O_PRI1,  512,  512,  512,  0);
  TRS(pri_w2,   O_PRI2,  512,  512,  512,  0);
  TRS(pri_mu_w, O_PRIMU, 512,  64,   512,  0);
  TRS(pri_lv_w, O_PRILV, 512,  64,   512,  0);
  TRS(gen_w1,   O_GEN1,  1024, 512,  1024, 0);
  TRS(gen_w2,   O_GEN2,  512,  512,  512,  0);
  TRS(gen_mu_w, O_GENMU, 512,  64,   512,  0);
  TRS(lstm_wih, O_LSTM,  1024, 2048, 1536, 0);
  TRS(lstm_whh, O_LSTM,  512,  2048, 1536, 1024);

  // Phase B: fx for all T*B rows (4096 WGs x 16 rows)
  fx_kernel<<<(TT * BB) / 16, 1024, 0, stream>>>(x, fx_b1, fx_b2, wsb, wsb + O_FXG);

  // Phase C: persistent recurrence (16 WGs x 16 batch rows, 32 waves each)
  vrnn_step_kernel<<<BB / 16, 1024, 0, stream>>>(
      eps, inf_b1, inf_b2, inf_mu_b, inf_lv_b, fz_b1, fz_b2,
      pri_b1, pri_b2, pri_mu_b, pri_lv_b, gen_b1, gen_b2, gen_mu_b,
      lstm_bih, lstm_bhh, wsb, cws, (float*)d_out);
}
